// MultiScaleGaussianFilter_60524679135683
// MI455X (gfx1250) — matrix operand order repaired and checked by reading, not compiled, not run on hardware
//
#include <hip/hip_runtime.h>

// ---------------- problem constants ----------------
#define B_   8
#define C_   64
#define H_   256
#define W_   256
#define CK   192          // 3 * 64 concat channels (GEMM K)
#define TW   32           // tile width
#define TH   8            // tile height
#define RMAX 6            // max blur radius (sigma=2 -> k=13)
#define XHALO 8           // x halo widened to 8 so async b128 segments are 16B aligned
#define PW   (TW + 2*XHALO)  // 48
#define PH   (TH + 2*RMAX)   // 20
#define CCH  16           // channels per blur chunk
#define NPIX (TW*TH)      // 256 pixels per tile
#define NTHREADS 256
#define OUT_STRIDE 268    // padded f32 stride for D transpose (conflict-free)

typedef __bf16 bf16_t;
typedef __attribute__((ext_vector_type(16))) __bf16 v16bf;
typedef __attribute__((ext_vector_type(8)))  float  v8f;
typedef __attribute__((ext_vector_type(2)))  float  v2f;
typedef __attribute__((ext_vector_type(4)))  unsigned int v4u;
typedef int v4i_vs __attribute__((vector_size(16)));   // matches async builtin pointee

#define AS1 __attribute__((address_space(1)))
#define AS3 __attribute__((address_space(3)))

#if defined(__has_builtin)
#if __has_builtin(__builtin_amdgcn_global_load_async_to_lds_b128)
#define HAVE_ASYNC_LDS 1
#else
#define HAVE_ASYNC_LDS 0
#endif
#else
#define HAVE_ASYNC_LDS 0
#endif

// ---------------- LDS layout (bytes) ----------------
constexpr size_t S_IN_BYTES = (size_t)CCH*PH*PW*4;              // 61440 per buffer
constexpr size_t OFF_IN0  = 0;
constexpr size_t OFF_IN1  = OFF_IN0 + S_IN_BYTES;               // 61440
constexpr size_t OFF_H    = OFF_IN1 + S_IN_BYTES;               // 122880: [CCH][PH][TW] f32
constexpr size_t OFF_OUT  = 0;                                  // [64][268] f32, aliases IN0+IN1
constexpr size_t OFF_W    = OFF_H    + (size_t)CCH*PH*TW*4;     // 163840: [64][192] bf16
constexpr size_t OFF_BIAS = OFF_W    + (size_t)C_*CK*2;         // 188416: [64] f32
constexpr size_t OFF_A    = OFF_BIAS + 256;                     // 188672: [256][192] bf16
constexpr size_t SMEM_TOTAL = OFF_A  + (size_t)NPIX*CK*2;       // 286976
static_assert(SMEM_TOTAL <= 320*1024, "over WGP LDS");
static_assert((size_t)C_*OUT_STRIDE*4 <= OFF_H, "s_out must fit in aliased region");

// ---------------- async helpers ----------------
template <unsigned short N>
__device__ __forceinline__ void wait_asynccnt() {
#if defined(__has_builtin) && __has_builtin(__builtin_amdgcn_s_wait_asynccnt)
  __builtin_amdgcn_s_wait_asynccnt(N);
#else
  asm volatile("s_wait_asynccnt %0" :: "n"(N) : "memory");
#endif
}

#if HAVE_ASYNC_LDS
// Copy one channel chunk (16 ch x 20 rows x 48 floats) global -> LDS via the
// async DMA path. 12 b128 segments per row; 3840 ops = 15 per thread exactly.
__device__ __forceinline__ void async_load_chunk(const float* __restrict__ gbase,
                                                 float* __restrict__ s_in, int tid) {
  for (int i = tid; i < CCH*PH*12; i += NTHREADS) {
    const int cl  = i / (PH*12);
    const int rem = i - cl*(PH*12);
    const int row = rem / 12;
    const int seg = rem - row*12;
    const float* g = gbase + ((size_t)cl*H_ + row)*W_ + seg*4;
    float*       l = s_in + (cl*PH + row)*PW + seg*4;
    __builtin_amdgcn_global_load_async_to_lds_b128(
        (AS1 v4i_vs*)(uintptr_t)g,
        (AS3 v4i_vs*)(unsigned int)(uintptr_t)l,
        0, 0);
  }
}
#endif

// ---------------- normalized separable Gaussian taps ----------------
template<int R> struct Taps;
template<> struct Taps<2> {  // sigma = 0.5, k = 5
  static constexpr float t[5] = {
    0.00026386922f, 0.10645077f, 0.78657072f, 0.10645077f, 0.00026386922f };
};
template<> struct Taps<3> {  // sigma = 1.0, k = 7
  static constexpr float t[7] = {
    0.0044330484f, 0.0540055826f, 0.2420362295f, 0.3990502790f,
    0.2420362295f, 0.0540055826f, 0.0044330484f };
};
template<> struct Taps<6> {  // sigma = 2.0, k = 13
  static constexpr float t[13] = {
    0.0022181945f, 0.0087731299f, 0.0270231640f, 0.0648252112f,
    0.1211093077f, 0.1762130845f, 0.1996756286f, 0.1762130845f,
    0.1211093077f, 0.0648252112f, 0.0270231640f, 0.0087731299f,
    0.0022181945f };
};

// one scale: horizontal pass (s_in -> s_h, row pairs), vertical pass
// (s_h -> bf16 A matrix, column pairs). Both use 2-wide packed f32 FMA.
template<int R>
__device__ __forceinline__ void blur_scale(const float* __restrict__ s_in,
                                           float* __restrict__ s_h,
                                           bf16_t* __restrict__ s_a,
                                           int sIdx, int cbase, int tid) {
  constexpr int K    = 2*R + 1;
  constexpr int ROWS = TH + 2*R;                 // rows actually needed: 12/14/20 (even)
  constexpr int HPAIRS = CCH*(ROWS/2)*TW;        // 3072 / 3584 / 5120 (x256 exact)
  __syncthreads();                               // s_in ready / s_h reusable
  // horizontal: pairs of rows, packed FMA; per-row reads stay consecutive (b128-able)
  for (int i = tid; i < HPAIRS; i += NTHREADS) {
    const int cl  = i / ((ROWS/2)*TW);
    const int rem = i - cl*((ROWS/2)*TW);
    const int pr  = rem >> 5;                    // row-pair index
    const int px  = rem & (TW-1);
    const int py  = (RMAX - R) + 2*pr;
    const float* s0 = &s_in[(cl*PH + py)*PW + px + (XHALO - R)];
    const float* s1 = s0 + PW;
    v2f acc = {0.f, 0.f};
#pragma unroll
    for (int j = 0; j < K; ++j) {
      const v2f t = {Taps<R>::t[j], Taps<R>::t[j]};
      const v2f v = {s0[j], s1[j]};
      acc = __builtin_elementwise_fma(t, v, acc);
    }
    float* d = &s_h[(cl*PH + py)*TW + px];
    d[0]  = acc.x;
    d[TW] = acc.y;
  }
  __syncthreads();                               // s_h ready
  // vertical: pairs of columns -> aligned b64 loads + packed FMA -> bf16 A[pix][k]
  for (int i = tid; i < CCH*(NPIX/2); i += NTHREADS) {   // 2048 = 8*256 exact
    const int cl = i >> 7;                       // / (NPIX/2)
    const int p2 = i & 127;
    const int ty = p2 >> 4;                      // / (TW/2)
    const int tx = (p2 & 15) * 2;
    const float* src = &s_h[(cl*PH + ty + (RMAX - R))*TW + tx];   // 8B aligned
    v2f acc = {0.f, 0.f};
#pragma unroll
    for (int j = 0; j < K; ++j) {
      const v2f t = {Taps<R>::t[j], Taps<R>::t[j]};
      const v2f v = *(const v2f*)&src[j*TW];
      acc = __builtin_elementwise_fma(t, v, acc);
    }
    const int pix = ty*TW + tx;
    const int k   = sIdx*C_ + cbase + cl;
    s_a[(size_t)pix*CK + k]     = (bf16_t)acc.x;
    s_a[(size_t)(pix+1)*CK + k] = (bf16_t)acc.y;
  }
}

extern "C" __global__ __launch_bounds__(NTHREADS)
void msgf_fused_kernel(const float* __restrict__ x,
                       const float* __restrict__ fusion_w,
                       const float* __restrict__ fusion_b,
                       float* __restrict__ out) {
  extern __shared__ char smem[];
  float*  s_in0  = (float*)(smem + OFF_IN0);
  float*  s_in1  = (float*)(smem + OFF_IN1);
  float*  s_h    = (float*)(smem + OFF_H);
  bf16_t* s_w    = (bf16_t*)(smem + OFF_W);
  float*  s_bias = (float*)(smem + OFF_BIAS);
  bf16_t* s_a    = (bf16_t*)(smem + OFF_A);
  float*  s_out  = (float*)(smem + OFF_OUT);     // aliases s_in0/1 (dead after blur)

  const int tid = threadIdx.x;
  const int x0  = blockIdx.x * TW;
  const int y0  = blockIdx.y * TH;
  const int b   = blockIdx.z;

  // weights (bf16, B layout: n-major rows of K) + bias into LDS
  for (int i = tid; i < C_*CK; i += NTHREADS) s_w[i] = (bf16_t)fusion_w[i];  // 48*256 exact
  if (tid < C_) s_bias[tid] = fusion_b[tid];

#if HAVE_ASYNC_LDS
  const bool interior = (x0 >= XHALO) && (x0 + TW + XHALO <= W_) &&
                        (y0 >= RMAX)  && (y0 + TH + RMAX  <= H_);
  const float* gtile = x + (((size_t)b*C_)*H_ + (y0 - RMAX))*W_ + (x0 - XHALO);
  if (interior) async_load_chunk(gtile, s_in0, tid);     // pipeline prologue: chunk 0
#endif

  // ---- stage 1: multi-scale separable blur -> bf16 A matrix [256 pix][192 k] ----
  for (int cc = 0; cc < C_/CCH; ++cc) {
    float* bufc = (cc & 1) ? s_in1 : s_in0;
#if HAVE_ASYNC_LDS
    if (interior) {
      // issue next chunk's DMA into the alternate buffer, then wait only for the
      // current chunk (15 b128 issued per thread per chunk; in-order completion)
      if (cc + 1 < C_/CCH) {
        float* bufn = (cc & 1) ? s_in0 : s_in1;
        async_load_chunk(gtile + (size_t)(cc+1)*CCH*H_*W_, bufn, tid);
        wait_asynccnt<15>();
      } else {
        wait_asynccnt<0>();
      }
    } else
#endif
    {
      __syncthreads();                 // previous reads of bufc complete
      for (int i = tid; i < CCH*PH*PW; i += NTHREADS) {  // 15360 = 60*256 exact
        const int cl  = i / (PH*PW);
        const int rem = i - cl*(PH*PW);
        const int py  = rem / PW;
        const int px  = rem - py*PW;
        const int gy  = y0 + py - RMAX;
        const int gx  = x0 + px - XHALO;
        float v = 0.f;                 // "SAME" zero padding
        if (gy >= 0 && gy < H_ && gx >= 0 && gx < W_)
          v = x[(((size_t)b*C_ + (cc*CCH + cl))*H_ + gy)*W_ + gx];
        bufc[i] = v;
      }
    }
    blur_scale<2>(bufc, s_h, s_a, 0, cc*CCH, tid);   // leading barrier makes DMA visible
    blur_scale<3>(bufc, s_h, s_a, 1, cc*CCH, tid);
    blur_scale<6>(bufc, s_h, s_a, 2, cc*CCH, tid);
  }
  __syncthreads();                       // A + W complete; s_in/s_h now dead

  // ---- stage 2: 1x1 fusion GEMM via bf16 WMMA, f32 accumulate ----
  const int lane = tid & 31;
  const int wv   = tid >> 5;             // wave 0..7
  const int nt   = wv & 3;               // N-tile (16 out channels)
  const int mg   = wv >> 2;              // M-group (8 M-tiles each)
  const int l16  = lane & 15;
  const int hf   = lane >> 4;            // lane half selects K sub-block

  // B fragments: lane l16 -> column n, halves hold K 0-15 / 16-31 of each 32-chunk
  v16bf Bm[6];
  {
    const int n = nt*16 + l16;
#pragma unroll
    for (int kk = 0; kk < 6; ++kk) {
      const bf16_t* p = &s_w[n*CK + kk*32 + hf*16];
      union { v4u u[2]; v16bf v; } t;
      t.u[0] = *(const v4u*)p;
      t.u[1] = *(const v4u*)(p + 8);
      Bm[kk] = t.v;
    }
  }

  const v8f vzero = {0.f,0.f,0.f,0.f,0.f,0.f,0.f,0.f};
  v8f acc[8];
#pragma unroll
  for (int m = 0; m < 8; ++m) acc[m] = vzero;

#pragma unroll
  for (int m = 0; m < 8; ++m) {
    const int pix = (mg*8 + m)*16 + l16;       // A row M = l16
#pragma unroll
    for (int kk = 0; kk < 6; ++kk) {
      // A fragment: K = kk*32 + (hf?16:0)+{0..15}
      const bf16_t* p = &s_a[(size_t)pix*CK + kk*32 + hf*16];
      union { v4u u[2]; v16bf v; } t;
      t.u[0] = *(const v4u*)p;
      t.u[1] = *(const v4u*)(p + 8);
      acc[m] = __builtin_amdgcn_wmma_f32_16x16x32_bf16(
          /*neg_a=*/false, t.v, /*neg_b=*/false, Bm[kk],
          /*c_mod=*/(short)0, acc[m], /*reuse_a=*/false, /*reuse_b=*/false);
    }
  }

  // D layout: lane l, vgpr r -> M = r + 8*(l>=16), N = l%16. Transpose via LDS.
#pragma unroll
  for (int m = 0; m < 8; ++m) {
    const int o = nt*16 + l16;
#pragma unroll
    for (int rr = 0; rr < 8; ++rr) {
      const int p = (mg*8 + m)*16 + rr + hf*8;
      s_out[o*OUT_STRIDE + p] = acc[m][rr];
    }
  }
  __syncthreads();                       // s_out complete

  // ---- stage 3: bias + residual + coalesced float4 store ----
#pragma unroll
  for (int jj = 0; jj < 16; ++jj) {
    const int flat = tid + jj*NTHREADS;  // 0..4095 (64 ch * 64 float4)
    const int o    = flat >> 6;
    const int rem  = flat & 63;
    const int p4   = rem * 4;            // pixel start (16B aligned in s_out)
    const int py   = p4 >> 5;
    const int px   = p4 & (TW-1);
    const size_t gidx = (((size_t)b*C_ + o)*H_ + (y0 + py))*W_ + (x0 + px);
    const float4 r4 = *(const float4*)&x[gidx];
    const float4 s4 = *(const float4*)&s_out[o*OUT_STRIDE + p4];
    const float bo  = s_bias[o];
    float4 v;
    v.x = r4.x + bo + s4.x;
    v.y = r4.y + bo + s4.y;
    v.z = r4.z + bo + s4.z;
    v.w = r4.w + bo + s4.w;
    *(float4*)&out[gidx] = v;
  }
}

extern "C" void kernel_launch(void* const* d_in, const int* in_sizes, int n_in,
                              void* d_out, int out_size, void* d_ws, size_t ws_size,
                              hipStream_t stream) {
  const float* x        = (const float*)d_in[0];
  const float* fusion_w = (const float*)d_in[1];
  const float* fusion_b = (const float*)d_in[2];
  float* out = (float*)d_out;

  // allow >64KB dynamic LDS (gfx1250 WGP has 320KB)
  (void)hipFuncSetAttribute((const void*)msgf_fused_kernel,
                            hipFuncAttributeMaxDynamicSharedMemorySize,
                            (int)SMEM_TOTAL);

  dim3 grid(W_/TW, H_/TH, B_);           // 8 x 32 x 8 = 2048 tiles
  msgf_fused_kernel<<<grid, dim3(NTHREADS), SMEM_TOTAL, stream>>>(x, fusion_w, fusion_b, out);
}
